// FNO2d_8770323218493
// MI455X (gfx1250) — matrix operand order repaired and checked by reading, not compiled, not run on hardware
//
#include <hip/hip_runtime.h>
#include <math.h>
#include <stdint.h>

// ---------------------------------------------------------------------------
// FNO2d for MI455X (gfx1250): every heavy op is a GEMM on
// v_wmma_f32_16x16x32_bf16.  2D rFFT/irFFT realized as DFT matmuls with
// precomputed twiddle matrices.  LDS tiles are staged with CDNA5 async
// global->LDS copies (ASYNCcnt) and double-buffered so the next K-slab
// streams in while the current one feeds the WMMAs.
// ---------------------------------------------------------------------------

typedef __attribute__((ext_vector_type(16))) __bf16 v16bf;
typedef __attribute__((ext_vector_type(8)))  float  v8f;
typedef __attribute__((ext_vector_type(4)))  float  v4f;

#define FLAG_ACCUM 1   // D = D + alpha*(A@B)  (else D = alpha*(A@B))
#define FLAG_GELU  2   // final exact-erf gelu on result

__device__ __forceinline__ float gelu_exact(float x) {
    return 0.5f * x * (1.0f + erff(x * 0.70710678118654752440f));
}

__device__ __forceinline__ uint32_t lds_off32(const void* p) {
    // Generic pointers to LDS carry the LDS byte offset in the low 32 bits
    // (flat aperture layout, ISA 10.2).
    return (uint32_t)(uintptr_t)p;
}

__device__ __forceinline__ void async_copy_b128(uint32_t lds, const float* g) {
    uint64_t ga = (uint64_t)(uintptr_t)g;
    asm volatile("global_load_async_to_lds_b128 %0, %1, off"
                 :: "v"(lds), "v"(ga) : "memory");
}
__device__ __forceinline__ void async_copy_b32(uint32_t lds, const float* g) {
    uint64_t ga = (uint64_t)(uintptr_t)g;
    asm volatile("global_load_async_to_lds_b32 %0, %1, off"
                 :: "v"(lds), "v"(ga) : "memory");
}

// ---------------------------------------------------------------------------
// Twiddle matrices (argument reduced mod 256 so cosf/sinf stay accurate):
//   cwf/swf : [129][256]  forward W rDFT   (swf = -sin)
//   cif/sif : [129][256]  inverse W rDFT, Hermitian-weighted, /256 folded in
//   chf/shf : [256][256]  H-direction DFT (symmetric; reused for inverse)
// ---------------------------------------------------------------------------
__global__ void fill_twiddles(float* __restrict__ cwf, float* __restrict__ swf,
                              float* __restrict__ cif, float* __restrict__ sif,
                              float* __restrict__ chf, float* __restrict__ shf)
{
    int t = blockIdx.x * blockDim.x + threadIdx.x;
    if (t >= 256 * 256) return;
    int m = t >> 8, h = t & 255;
    int ph = (m * h) & 255;                        // exact periodic reduction
    float ang = (float)ph * (6.2831853071795864769f / 256.0f);
    float c = cosf(ang), s = sinf(ang);
    chf[t] = c;
    shf[t] = s;
    if (m < 129) {
        int idx = m * 256 + h;
        cwf[idx] = c;
        swf[idx] = -s;                             // rfft imag part sign
        float ak = (m == 0 || m == 128) ? 1.0f : 2.0f;   // Hermitian weight
        cif[idx] =  ak * c * (1.0f / 256.0f);
        sif[idx] = -ak * s * (1.0f / 256.0f);
    }
}

// ---------------------------------------------------------------------------
// Generic batched strided GEMM (bf16 WMMA, f32 accumulate):
//   D[bz][m][n] (+)= alpha * sum_k A[bz][m][k]*B[bz][k][n] (+bias[m]) (+Add) (gelu)
// A element (m,k): A + bz*aBatch + m*aRS + k*aCS
// B element (k,n): B + bz*bBatch + k*bRS + n*bCS
// D row-major [m][n], row stride N, batch stride dBatch.
// Tile 64x128x32; 256 threads = 8 waves (2 along M x 4 along N);
// each wave owns a 32x32 slab -> 4 WMMAs per k-step (fragment reuse 2x2).
// Double-buffered LDS, async-pipelined staging on interior blocks.
// Template: AKC = (aCS==1) k-contiguous A;  BKC = (bRS==1) k-contiguous B.
// ---------------------------------------------------------------------------
template <bool AKC, bool BKC>
__global__ __launch_bounds__(256) void gemm_bf16_wmma(
    const float* __restrict__ A, long long aBatch, long long aRS, long long aCS,
    const float* __restrict__ Bm, long long bBatch, long long bRS, long long bCS,
    float* __restrict__ D, long long dBatch,
    const float* __restrict__ Add, long long addBatch,
    const float* __restrict__ bias,
    int M, int N, int K, float alpha, int flags)
{
    __shared__ __align__(16) float As[2][64][40];    // [buf][m][k], row 160B
    __shared__ __align__(16) float Bs[2][128][40];   // [buf][n][k] (transposed)

    const int tid  = threadIdx.x;
    const int lane = tid & 31;
    const int wave = tid >> 5;
    const int wm   = wave >> 2;          // 0..1 : 32-row slab
    const int wn   = wave & 3;           // 0..3 : 32-col slab

    const int m0 = blockIdx.y * 64;
    const int n0 = blockIdx.x * 128;
    const int bz = blockIdx.z;

    const float* Ab = A  + (long long)bz * aBatch;
    const float* Bb = Bm + (long long)bz * bBatch;
    float*       Db = D  + (long long)bz * dBatch;

    const bool fullM = (m0 + 64  <= M);
    const bool fullN = (n0 + 128 <= N);
    const bool alignedA = AKC && ((aRS & 3) == 0);   // 16B rows for b128
    const bool alignedB = BKC && ((bCS & 3) == 0);

    v8f acc[2][2] = {};

    // ---------------- staging helpers ----------------
    auto stageA = [&](int k0, int buf, bool fast) {
        if (fast) {
            if (AKC) {
                if (alignedA) {
                    #pragma unroll
                    for (int j = 0; j < 2; ++j) {          // 512 x 16B chunks
                        int c   = tid + j * 256;
                        int row = c >> 3, k4 = (c & 7) << 2;
                        const float* g = Ab + (long long)(m0 + row) * aRS + (k0 + k4);
                        async_copy_b128(lds_off32(&As[buf][row][k4]), g);
                    }
                } else {                                    // 4B rows: b32 along k
                    #pragma unroll
                    for (int j = 0; j < 8; ++j) {
                        int e   = tid + j * 256;
                        int row = e >> 5, kk = e & 31;
                        const float* g = Ab + (long long)(m0 + row) * aRS + (k0 + kk);
                        async_copy_b32(lds_off32(&As[buf][row][kk]), g);
                    }
                }
            } else {                                        // m-contiguous source
                #pragma unroll
                for (int j = 0; j < 8; ++j) {
                    int e  = tid + j * 256;
                    int mm = e & 63, kk = e >> 6;
                    const float* g = Ab + (long long)(m0 + mm) * aRS
                                        + (long long)(k0 + kk) * aCS;
                    async_copy_b32(lds_off32(&As[buf][mm][kk]), g);
                }
            }
        } else {
            for (int i = tid; i < 64 * 32; i += 256) {
                int mm = i >> 5, kk = i & 31;
                int gm = m0 + mm, gk = k0 + kk;
                float v = 0.0f;
                if (gm < M && gk < K)
                    v = Ab[(long long)gm * aRS + (long long)gk * aCS];
                As[buf][mm][kk] = v;
            }
        }
    };

    auto stageB = [&](int k0, int buf, bool fast) {
        if (fast) {
            if (BKC) {
                if (alignedB) {
                    #pragma unroll
                    for (int j = 0; j < 4; ++j) {          // 1024 x 16B chunks
                        int c  = tid + j * 256;
                        int nn = c >> 3, k4 = (c & 7) << 2;
                        const float* g = Bb + (long long)(k0 + k4)
                                            + (long long)(n0 + nn) * bCS;
                        async_copy_b128(lds_off32(&Bs[buf][nn][k4]), g);
                    }
                } else {
                    #pragma unroll
                    for (int j = 0; j < 16; ++j) {
                        int e  = tid + j * 256;
                        int nn = e >> 5, kk = e & 31;
                        const float* g = Bb + (long long)(k0 + kk)
                                            + (long long)(n0 + nn) * bCS;
                        async_copy_b32(lds_off32(&Bs[buf][nn][kk]), g);
                    }
                }
            } else {                                        // n-contiguous source
                #pragma unroll
                for (int j = 0; j < 16; ++j) {
                    int e  = tid + j * 256;
                    int nn = e & 127, kk = e >> 7;
                    const float* g = Bb + (long long)(k0 + kk) * bRS
                                        + (long long)(n0 + nn) * bCS;
                    async_copy_b32(lds_off32(&Bs[buf][nn][kk]), g);
                }
            }
        } else {
            for (int i = tid; i < 128 * 32; i += 256) {
                int nn = i >> 5, kk = i & 31;
                int gn = n0 + nn, gk = k0 + kk;
                float v = 0.0f;
                if (gk < K && gn < N)
                    v = Bb[(long long)gk * bRS + (long long)gn * bCS];
                Bs[buf][nn][kk] = v;
            }
        }
    };

    // ---- fragment gather + 4 WMMAs (ISA 7.12.2 layouts, wave32) ----
    auto compute = [&](int buf) {
        const int rr = lane & 15;
        const int ka = (lane < 16) ? 0 : 8;   // A: lanes16-31 hold K 8..15/24..31
        const int kb = (lane < 16) ? 0 : 8;   // B: lanes16-31 hold K 8..15/24..31
        v16bf af[2], bfr[2];
        #pragma unroll
        for (int t = 0; t < 2; ++t) {
            const v4f* arow = (const v4f*)&As[buf][wm * 32 + t * 16 + rr][0];
            v4f a0 = arow[(ka >> 2) + 0];
            v4f a1 = arow[(ka >> 2) + 1];
            v4f a2 = arow[(ka >> 2) + 4];
            v4f a3 = arow[(ka >> 2) + 5];
            #pragma unroll
            for (int e = 0; e < 4; ++e) {
                af[t][e]      = (__bf16)a0[e];
                af[t][e + 4]  = (__bf16)a1[e];
                af[t][e + 8]  = (__bf16)a2[e];
                af[t][e + 12] = (__bf16)a3[e];
            }
        }
        #pragma unroll
        for (int u = 0; u < 2; ++u) {
            const v4f* brow = (const v4f*)&Bs[buf][wn * 32 + u * 16 + rr][0];
            #pragma unroll
            for (int q = 0; q < 4; ++q) {
                v4f b = brow[(kb >> 2) + (q & 1) + (q >> 1) * 4];
                #pragma unroll
                for (int e = 0; e < 4; ++e)
                    bfr[u][q * 4 + e] = (__bf16)b[e];
            }
        }
        #pragma unroll
        for (int t = 0; t < 2; ++t)
            #pragma unroll
            for (int u = 0; u < 2; ++u)
                acc[t][u] = __builtin_amdgcn_wmma_f32_16x16x32_bf16(
                    false, af[t], false, bfr[u], (short)0, acc[t][u], false, false);
    };

    // ---------------- main loop ----------------
    const bool uniform = fullM && fullN && ((K & 31) == 0) &&
                         (!AKC || alignedA) && (!BKC || alignedB);
    if (uniform) {
        // per-wave async ops per k-step (compile-time for the wait immediate)
        constexpr int NOPS = (AKC ? 2 : 8) + (BKC ? 4 : 16);
        const int nsteps = K >> 5;
        stageA(0, 0, true);
        stageB(0, 0, true);
        for (int s = 0; s < nsteps; ++s) {
            const int cur = s & 1;
            if (s + 1 < nsteps) {
                stageA((s + 1) << 5, 1 - cur, true);
                stageB((s + 1) << 5, 1 - cur, true);
                // newest batch may stay in flight; batch s must be complete
                asm volatile("s_wait_asynccnt %0" :: "i"(NOPS) : "memory");
            } else {
                asm volatile("s_wait_asynccnt 0x0" ::: "memory");
            }
            __syncthreads();
            compute(cur);
            __syncthreads();
        }
    } else {
        for (int k0 = 0; k0 < K; k0 += 32) {
            const bool fullK = (k0 + 32 <= K);
            stageA(k0, 0, fullM && fullK);
            stageB(k0, 0, fullN && fullK);
            asm volatile("s_wait_asynccnt 0x0" ::: "memory");
            __syncthreads();
            compute(0);
            __syncthreads();
        }
    }

    // ---- epilogue: C/D layout = 8 rows per half-wave, col = lane&15 ----
    const int rr = lane & 15;
    const float* AddB = Add ? (Add + (long long)bz * addBatch) : nullptr;
    #pragma unroll
    for (int t = 0; t < 2; ++t) {
        const int rbase = m0 + wm * 32 + t * 16 + ((lane < 16) ? 0 : 8);
        #pragma unroll
        for (int u = 0; u < 2; ++u) {
            const int col = n0 + wn * 32 + u * 16 + rr;
            if (col >= N) continue;
            #pragma unroll
            for (int v = 0; v < 8; ++v) {
                int gm = rbase + v;
                if (gm < M) {
                    long long idx = (long long)gm * N + col;
                    float r = acc[t][u][v] * alpha;
                    if (flags & FLAG_ACCUM) r += Db[idx];
                    if (bias)               r += bias[gm];
                    if (AddB)               r += AddB[idx];
                    if (flags & FLAG_GELU)  r = gelu_exact(r);
                    Db[idx] = r;
                }
            }
        }
    }
}

// ---------------------------------------------------------------------------
// Host-side orchestration
// ---------------------------------------------------------------------------
extern "C" void kernel_launch(void* const* d_in, const int* in_sizes, int n_in,
                              void* d_out, int out_size, void* d_ws, size_t ws_size,
                              hipStream_t stream)
{
    (void)in_sizes; (void)n_in; (void)out_size; (void)ws_size;

    const float* x        = (const float*)d_in[0];
    const float* lin_in_w = (const float*)d_in[1];
    const float* lin_in_b = (const float*)d_in[2];
    const float* wr       = (const float*)d_in[3];
    const float* wi       = (const float*)d_in[4];
    const float* br       = (const float*)d_in[5];
    const float* bi       = (const float*)d_in[6];
    const float* cw       = (const float*)d_in[7];
    const float* cb       = (const float*)d_in[8];
    const float* lout_w   = (const float*)d_in[9];
    const float* lout_b   = (const float*)d_in[10];
    float* out = (float*)d_out;

    constexpr int  Bn = 4, CIN = 3, COUT = 1, HID = 128, LNUM = 4;
    constexpr int  Hh = 256, Ww = 256, WF = 129;
    constexpr long long HW = (long long)Hh * Ww;   // 65536
    constexpr long long NF = (long long)Hh * WF;   // 33024

    // ---- carve workspace (256B aligned chunks) ----
    char* p = (char*)d_ws;
    auto carve = [&](size_t bytes) -> float* {
        char* r = p;
        p += (bytes + 255) & ~(size_t)255;
        return (float*)r;
    };
    float* twc_w = carve((size_t)WF * 256 * 4);
    float* tws_w = carve((size_t)WF * 256 * 4);
    float* twc_i = carve((size_t)WF * 256 * 4);
    float* tws_i = carve((size_t)WF * 256 * 4);
    float* twc_h = carve((size_t)256 * 256 * 4);
    float* tws_h = carve((size_t)256 * 256 * 4);
    float* hbuf  = carve((size_t)Bn * HID * HW * 4);   // activations (B,HID,H,W)
    float* spat  = carve((size_t)Bn * HID * HW * 4);   // 1x1-conv branch
    float* fa_re = carve((size_t)Bn * HID * NF * 4);   // spectral ping re
    float* fa_im = carve((size_t)Bn * HID * NF * 4);   // spectral ping im
    float* fb_re = carve((size_t)Bn * HID * NF * 4);   // spectral pong re
    float* fb_im = carve((size_t)Bn * HID * NF * 4);   // spectral pong im

    fill_twiddles<<<dim3((256 * 256 + 255) / 256), dim3(256), 0, stream>>>(
        twc_w, tws_w, twc_i, tws_i, twc_h, tws_h);

    auto gemm = [&](const float* A, long long aB, long long aRS, long long aCS,
                    const float* Bp, long long bB, long long bRS, long long bCS,
                    float* Dp, long long dB,
                    const float* Addp, long long addB,
                    const float* biasp,
                    int M, int N, int K, int z, float alpha, int flags) {
        dim3 g((unsigned)((N + 127) / 128), (unsigned)((M + 63) / 64), (unsigned)z);
        dim3 b(256);
        const bool akc = (aCS == 1);
        const bool bkc = (bRS == 1);
        if (akc && bkc)
            gemm_bf16_wmma<true, true><<<g, b, 0, stream>>>(
                A, aB, aRS, aCS, Bp, bB, bRS, bCS, Dp, dB, Addp, addB, biasp, M, N, K, alpha, flags);
        else if (akc && !bkc)
            gemm_bf16_wmma<true, false><<<g, b, 0, stream>>>(
                A, aB, aRS, aCS, Bp, bB, bRS, bCS, Dp, dB, Addp, addB, biasp, M, N, K, alpha, flags);
        else if (!akc && bkc)
            gemm_bf16_wmma<false, true><<<g, b, 0, stream>>>(
                A, aB, aRS, aCS, Bp, bB, bRS, bCS, Dp, dB, Addp, addB, biasp, M, N, K, alpha, flags);
        else
            gemm_bf16_wmma<false, false><<<g, b, 0, stream>>>(
                A, aB, aRS, aCS, Bp, bB, bRS, bCS, Dp, dB, Addp, addB, biasp, M, N, K, alpha, flags);
    };

    // ---- input transform: h = gelu(lin_in_w^T @ x + b), per batch ----
    gemm(lin_in_w, 0, 1, HID,
         x, (long long)CIN * HW, HW, 1,
         hbuf, (long long)HID * HW, nullptr, 0, lin_in_b,
         HID, (int)HW, CIN, Bn, 1.0f, FLAG_GELU);

    for (int l = 0; l < LNUM; ++l) {
        const float* wr_l = wr + (size_t)l * HID * HID;
        const float* wi_l = wi + (size_t)l * HID * HID;
        const float* br_l = br + (size_t)l * HID;
        const float* bi_l = bi + (size_t)l * HID;
        const float* cw_l = cw + (size_t)l * HID * HID;
        const float* cb_l = cb + (size_t)l * HID;

        // ---- spatial 1x1 conv: spat[o] = cw[o,i] @ h[i] + cb ----
        gemm(cw_l, 0, HID, 1,
             hbuf, (long long)HID * HW, HW, 1,
             spat, (long long)HID * HW, nullptr, 0, cb_l,
             HID, (int)HW, HID, Bn, 1.0f, 0);

        // ---- forward rDFT over W: Y = X @ Cw^T, Z = X @ Sw^T ----
        gemm(hbuf, HW, Ww, 1, twc_w, 0, 1, 256, fa_re, NF, nullptr, 0, nullptr,
             Hh, WF, Ww, Bn * HID, 1.0f, 0);
        gemm(hbuf, HW, Ww, 1, tws_w, 0, 1, 256, fa_im, NF, nullptr, 0, nullptr,
             Hh, WF, Ww, Bn * HID, 1.0f, 0);

        // ---- forward DFT over H: Re = C@Y + S@Z ; Im = C@Z - S@Y ----
        gemm(twc_h, 0, 256, 1, fa_re, NF, WF, 1, fb_re, NF, nullptr, 0, nullptr,
             Hh, WF, Hh, Bn * HID,  1.0f, 0);
        gemm(tws_h, 0, 256, 1, fa_im, NF, WF, 1, fb_re, NF, nullptr, 0, nullptr,
             Hh, WF, Hh, Bn * HID,  1.0f, FLAG_ACCUM);
        gemm(twc_h, 0, 256, 1, fa_im, NF, WF, 1, fb_im, NF, nullptr, 0, nullptr,
             Hh, WF, Hh, Bn * HID,  1.0f, 0);
        gemm(tws_h, 0, 256, 1, fa_re, NF, WF, 1, fb_im, NF, nullptr, 0, nullptr,
             Hh, WF, Hh, Bn * HID, -1.0f, FLAG_ACCUM);

        // ---- per-frequency channel mix: Re' = wr^T@Re + br ; Im' = wi^T@Im + bi ----
        gemm(wr_l, 0, 1, HID, fb_re, (long long)HID * NF, NF, 1,
             fa_re, (long long)HID * NF, nullptr, 0, br_l,
             HID, (int)NF, HID, Bn, 1.0f, 0);
        gemm(wi_l, 0, 1, HID, fb_im, (long long)HID * NF, NF, 1,
             fa_im, (long long)HID * NF, nullptr, 0, bi_l,
             HID, (int)NF, HID, Bn, 1.0f, 0);

        // ---- inverse DFT over H: Yre=(C@R - S@I)/256 ; Yim=(S@R + C@I)/256 ----
        gemm(twc_h, 0, 256, 1, fa_re, NF, WF, 1, fb_re, NF, nullptr, 0, nullptr,
             Hh, WF, Hh, Bn * HID,  1.0f / 256.0f, 0);
        gemm(tws_h, 0, 256, 1, fa_im, NF, WF, 1, fb_re, NF, nullptr, 0, nullptr,
             Hh, WF, Hh, Bn * HID, -1.0f / 256.0f, FLAG_ACCUM);
        gemm(tws_h, 0, 256, 1, fa_re, NF, WF, 1, fb_im, NF, nullptr, 0, nullptr,
             Hh, WF, Hh, Bn * HID,  1.0f / 256.0f, 0);
        gemm(twc_h, 0, 256, 1, fa_im, NF, WF, 1, fb_im, NF, nullptr, 0, nullptr,
             Hh, WF, Hh, Bn * HID,  1.0f / 256.0f, FLAG_ACCUM);

        // ---- inverse rDFT over W + fused epilogue: h = gelu(spat + Yre@Ci + Yim@Si) ----
        gemm(fb_re, NF, WF, 1, twc_i, 0, 256, 1, hbuf, HW, nullptr, 0, nullptr,
             Hh, Ww, WF, Bn * HID, 1.0f, 0);
        gemm(fb_im, NF, WF, 1, tws_i, 0, 256, 1, hbuf, HW, spat, HW, nullptr,
             Hh, Ww, WF, Bn * HID, 1.0f, FLAG_ACCUM | FLAG_GELU);
    }

    // ---- output transform: y = gelu(lin_out_w^T @ h + b) ----
    gemm(lout_w, 0, 1, COUT,
         hbuf, (long long)HID * HW, HW, 1,
         out, (long long)COUT * HW, nullptr, 0, lout_b,
         COUT, (int)HW, HID, Bn, 1.0f, FLAG_GELU);
}
